// GModel_90013924589674
// MI455X (gfx1250) — matrix operand order repaired and checked by reading, not compiled, not run on hardware
//
#include <hip/hip_runtime.h>
#include <hip/hip_bf16.h>
#include <math.h>

typedef __attribute__((ext_vector_type(16))) _Float16 v16h;
typedef __attribute__((ext_vector_type(8)))  _Float16 v8h;
typedef __attribute__((ext_vector_type(8)))  float    v8f;

__device__ __forceinline__ float leaky(float x) { return x > 0.f ? x : 0.2f * x; }

__device__ __forceinline__ v16h cat8(v8h lo, v8h hi) {
    return __builtin_shufflevector(lo, hi, 0,1,2,3,4,5,6,7,8,9,10,11,12,13,14,15);
}

// Async global->LDS DMA (CDNA5, tracked by ASYNCcnt). Generic LDS pointer's
// low 32 bits are the LDS byte offset (shared aperture layout).
__device__ __forceinline__ void async_copy_b128(void* lds_dst, const void* gsrc) {
    unsigned int      l = (unsigned int)(uintptr_t)lds_dst;
    unsigned long long g = (unsigned long long)(uintptr_t)gsrc;
    asm volatile("global_load_async_to_lds_b128 %0, %1, off"
                 :: "v"(l), "v"(g) : "memory");
}
__device__ __forceinline__ void wait_async_all() {
    asm volatile("s_wait_asynccnt 0" ::: "memory");
}

// ------------------------------------------------------------------
// Kernel 1: per-image  async-stage -> conv1(VALU) -> conv2(WMMA
// implicit GEMM, K reordered tap-major) -> GAP(ds_add_f32) -> proj.
// LDS map (bytes):
//   [0      .. 69696)  s_a1  [33][33][32ch] f16  (aliased: s_raw f32[8192] in stage phase)
//   [69696  .. 86596)  s_in  [65][65][2ch]  f16
//   [86608  .. 123472) s_w2k [9tap][64oc][32ic] f16  (16B aligned)
//   [123472 .. 126288) s_w1[576], s_c2b[64], s_gap[64]  f32
// ------------------------------------------------------------------
#define SA1_ELEMS (33*33*32)
#define SIN_ELEMS (65*65*2)
#define SW2_ELEMS (9*64*32)
#define SW2_OFF   86608
#define WF_OFF    123472
#define SMEM1_SZ  126288

__global__ __launch_bounds__(256) void k_convnet(
    const float* __restrict__ texts, const float* __restrict__ styles,
    const float* __restrict__ c1w, const float* __restrict__ c1b,
    const float* __restrict__ c2w, const float* __restrict__ c2b,
    const float* __restrict__ pw,  const float* __restrict__ pb,
    float* __restrict__ feats)
{
    extern __shared__ char smem[];
    _Float16* s_a1  = (_Float16*)smem;             // conv1 out, channel-last
    float*    s_raw = (float*)smem;                // staged f32 input (aliases s_a1)
    _Float16* s_in  = s_a1 + SA1_ELEMS;            // padded f16 input, channel-last
    _Float16* s_w2k = (_Float16*)(smem + SW2_OFF); // conv2 weights, [tap][oc][ic]
    float*    s_w1  = (float*)(smem + WF_OFF);
    float*    s_c2b = s_w1 + 576;
    float*    s_gap = s_c2b + 64;

    const int img = blockIdx.x;       // b*64 + t
    const int tid = threadIdx.x;

    // ---- phase A: async DMA of the 32KB f32 input; stage weights ---------
    for (int i4 = tid; i4 < 2048; i4 += 256) {
        const int i = i4 * 4;                      // float index, 16B aligned
        const float* src = (i < 4096) ? (texts  + (size_t)img*4096 + i)
                                      : (styles + (size_t)img*4096 + (i - 4096));
        async_copy_b128(s_raw + i, src);
    }
    // w2 permuted to [tap][oc][ic] so B-fragments are lane-contiguous
    for (int i = tid; i < SW2_ELEMS; i += 256) {
        const int tap = i >> 11;                   // i / 2048
        const int r   = i & 2047;
        const int oc  = r >> 5, ic = r & 31;
        s_w2k[i] = (_Float16)c2w[(oc*32 + ic)*9 + tap];
    }
    for (int i = tid; i < 576; i += 256) s_w1[i] = c1w[i];
    if (tid < 64) { s_c2b[tid] = c2b[tid]; s_gap[tid] = 0.f; }
    wait_async_all();
    __syncthreads();

    // ---- phase B: staged f32 -> padded channel-last f16 input ------------
    for (int i = tid; i < SIN_ELEMS; i += 256) {
        const int c = i & 1; const int p = i >> 1;
        const int y = p / 65, x = p - y*65;
        float v = (y < 64 && x < 64) ? s_raw[c*4096 + y*64 + x] : 0.f;
        s_in[p*2 + c] = (_Float16)v;
    }
    __syncthreads();                               // s_raw dead after this point

    // ---- phase C: zero a1 pads + conv1 (2->32ch, 3x3, s2, SAME, leaky) ---
    for (int i = tid; i < 65*32; i += 256) {       // pad row y=32 + col x=32
        const int p = i >> 5, c = i & 31;
        const int y = (p < 33) ? 32 : (p - 33);
        const int x = (p < 33) ? p  : 32;
        s_a1[(y*33 + x)*32 + c] = (_Float16)0.f;
    }
    for (int o = tid; o < 32*32*32; o += 256) {
        const int oc = o >> 10; const int pos = o & 1023;
        const int oy = pos >> 5, ox = pos & 31;
        float acc = c1b[oc];
        const float* w = s_w1 + oc*18;
        #pragma unroll
        for (int ky = 0; ky < 3; ++ky)
            #pragma unroll
            for (int kx = 0; kx < 3; ++kx) {
                const int p = (2*oy+ky)*65 + (2*ox+kx);
                acc += (float)s_in[p*2 + 0] * w[0*9 + ky*3 + kx];
                acc += (float)s_in[p*2 + 1] * w[1*9 + ky*3 + kx];
            }
        s_a1[(oy*33 + ox)*32 + oc] = (_Float16)leaky(acc);
    }
    __syncthreads();

    // ---- phase D: conv2 implicit GEMM, M=256 x N=64 x K=288 (tap-major) --
    // chunk kc == tap; chunk's 32 K-values == 32 channels at one position.
    const int wave = tid >> 5, lane = tid & 31;
    const int half = lane >> 4, lm = lane & 15;
    for (int mi = 0; mi < 2; ++mi) {
        const int mt = wave*2 + mi;          // M tile 0..15
        const int M  = mt*16 + lm;           // output position 0..255
        const int oy = M >> 4, ox = M & 15;
        v8f acc[4] = {{}, {}, {}, {}};
        for (int kc = 0; kc < 9; ++kc) {
            const int ky = kc / 3, kx = kc - 3*ky;
            // A fragment: two contiguous b128 (channels 8h..+7, 16+8h..+7)
            const v8h* ap = (const v8h*)(s_a1 + ((2*oy+ky)*33 + (2*ox+kx))*32);
            const v16h a = cat8(ap[half], ap[2 + half]);
            #pragma unroll
            for (int nt = 0; nt < 4; ++nt) {
                const int n = nt*16 + lm;           // out channel
                // B fragment: two b128 (ic 8h..+7, 16+8h..+7)
                const v8h* bp = (const v8h*)(s_w2k + (kc*64 + n)*32);
                const v16h bf = cat8(bp[half], bp[2 + half]);
                acc[nt] = __builtin_amdgcn_wmma_f32_16x16x32_f16(
                    false, a, false, bf, (short)0, acc[nt], false, false);
            }
        }
        #pragma unroll
        for (int nt = 0; nt < 4; ++nt) {
            const int n = nt*16 + lm;
            const float bias = s_c2b[n];
            float part = 0.f;
            #pragma unroll
            for (int r = 0; r < 8; ++r) part += leaky(acc[nt][r] + bias);
            atomicAdd(&s_gap[n], part);             // ds_add_f32
        }
    }
    __syncthreads();

    // ---- GAP scale (1/256) + projection 64->128 --------------------------
    if (tid < 128) {
        float acc = pb[tid];
        for (int j = 0; j < 64; ++j)
            acc += (s_gap[j] * (1.f/256.f)) * pw[j*128 + tid];
        feats[(size_t)img*128 + tid] = acc;
    }
}

// ------------------------------------------------------------------
// Kernel 2: per-batch-row scorer MLP + tanh + argmax + gather +
//           decoder layer-1 (128->512, leaky), h1 stored as f16.
// ------------------------------------------------------------------
__global__ __launch_bounds__(256) void k_score_route(
    const float* __restrict__ feats,
    const float* __restrict__ v1w, const float* __restrict__ v1b,
    const float* __restrict__ v2w, const float* __restrict__ v2b,
    const float* __restrict__ v3w, const float* __restrict__ v3b,
    const float* __restrict__ d1w, const float* __restrict__ d1b,
    _Float16* __restrict__ h1g)
{
    extern __shared__ char smem[];
    float* s_f  = (float*)smem;       // [64][128] feats (16B aligned)
    float* s_ha = s_f  + 8192;        // [64][128]
    float* s_hb = s_ha + 8192;        // [64][128]
    float* s_sc = s_hb + 8192;        // [64]
    __shared__ int s_idx;

    const int b = blockIdx.x, tid = threadIdx.x;
    for (int i4 = tid; i4 < 2048; i4 += 256)
        async_copy_b128(s_f + i4*4, feats + (size_t)b*8192 + i4*4);
    wait_async_all();
    __syncthreads();

    for (int o = tid; o < 8192; o += 256) {
        const int t = o >> 7, c = o & 127;
        float acc = v1b[c];
        for (int j = 0; j < 128; ++j) acc += s_f[t*128 + j] * v1w[j*128 + c];
        s_ha[o] = leaky(acc);
    }
    __syncthreads();
    for (int o = tid; o < 8192; o += 256) {
        const int t = o >> 7, c = o & 127;
        float acc = v2b[c];
        for (int j = 0; j < 128; ++j) acc += s_ha[t*128 + j] * v2w[j*128 + c];
        s_hb[o] = leaky(acc);
    }
    __syncthreads();
    if (tid < 64) {
        float acc = v3b[0];
        for (int j = 0; j < 128; ++j) acc += s_hb[tid*128 + j] * v3w[j];
        s_sc[tid] = tanhf(acc);
    }
    __syncthreads();
    if (tid == 0) {
        int best = 0; float bv = s_sc[0];
        for (int t = 1; t < 64; ++t) if (s_sc[t] > bv) { bv = s_sc[t]; best = t; }
        s_idx = best;                                  // first-max, like jnp.argmax
    }
    __syncthreads();
    for (int c = tid; c < 512; c += 256) {
        float acc = d1b[c];
        const float* fr = s_f + s_idx*128;
        for (int j = 0; j < 128; ++j) acc += fr[j] * d1w[j*512 + c];
        h1g[(size_t)b*512 + c] = (_Float16)leaky(acc);
    }
}

// ------------------------------------------------------------------
// Kernel 3: decoder GEMM [32x512] x [512x4096] via WMMA, fused
//           bias + tanh epilogue. One wave per 16x16 output tile.
// ------------------------------------------------------------------
__global__ __launch_bounds__(32) void k_decoder(
    const _Float16* __restrict__ h1g, const float* __restrict__ d2w,
    const float* __restrict__ d2b, float* __restrict__ out)
{
    const int nt = blockIdx.x, mt = blockIdx.y;     // 256 x 2 tiles
    const int lane = threadIdx.x, half = lane >> 4, lm = lane & 15;
    const int p = nt*16 + lm;                       // pixel column
    v8f acc = {};
    for (int kc = 0; kc < 16; ++kc) {
        v16h a, bf;
        const _Float16* arow = h1g + (size_t)(mt*16 + lm)*512 + kc*32;
        #pragma unroll
        for (int v = 0; v < 8; ++v) {
            const int kb = ((v < 4) ? 2*v : 16 + 2*(v-4)) + 8*half;
            a[2*v]   = arow[kb];
            a[2*v+1] = arow[kb+1];
            const int k0 = kc*32 + kb;
            bf[2*v]   = (_Float16)d2w[(size_t)k0*4096 + p];
            bf[2*v+1] = (_Float16)d2w[(size_t)(k0+1)*4096 + p];
        }
        acc = __builtin_amdgcn_wmma_f32_16x16x32_f16(
            false, a, false, bf, (short)0, acc, false, false);
    }
    const float bias = d2b[p];
    #pragma unroll
    for (int r = 0; r < 8; ++r) {
        const int row = mt*16 + half*8 + r;         // batch row 0..31
        out[(size_t)row*4096 + p] = tanhf(acc[r] + bias);
    }
}

// ------------------------------------------------------------------
extern "C" void kernel_launch(void* const* d_in, const int* in_sizes, int n_in,
                              void* d_out, int out_size, void* d_ws, size_t ws_size,
                              hipStream_t stream) {
    (void)in_sizes; (void)n_in; (void)out_size; (void)ws_size;
    const float* texts  = (const float*)d_in[0];
    const float* styles = (const float*)d_in[1];
    const float* c1w = (const float*)d_in[2];
    const float* c1b = (const float*)d_in[3];
    const float* c2w = (const float*)d_in[4];
    const float* c2b = (const float*)d_in[5];
    const float* pw  = (const float*)d_in[6];
    const float* pb  = (const float*)d_in[7];
    const float* v1w = (const float*)d_in[8];
    const float* v1b = (const float*)d_in[9];
    const float* v2w = (const float*)d_in[10];
    const float* v2b = (const float*)d_in[11];
    const float* v3w = (const float*)d_in[12];
    const float* v3b = (const float*)d_in[13];
    const float* d1w = (const float*)d_in[14];
    const float* d1b = (const float*)d_in[15];
    const float* d2w = (const float*)d_in[16];
    const float* d2b = (const float*)d_in[17];

    float*    feats = (float*)d_ws;                                  // [2048,128] f32
    _Float16* h1g   = (_Float16*)((char*)d_ws + (size_t)2048*128*4); // [32,512] f16

    k_convnet<<<2048, 256, SMEM1_SZ, stream>>>(texts, styles, c1w, c1b, c2w, c2b,
                                               pw, pb, feats);

    const size_t smem2 = (size_t)(3*8192 + 64)*sizeof(float);        // ~96 KB
    k_score_route<<<32, 256, smem2, stream>>>(feats, v1w, v1b, v2w, v2b,
                                              v3w, v3b, d1w, d1b, h1g);

    k_decoder<<<dim3(256, 2), 32, 0, stream>>>(h1g, d2w, d2b, (float*)d_out);
}
